// FAMModule_40561671143445
// MI455X (gfx1250) — matrix operand order repaired and checked by reading, not compiled, not run on hardware
//
#include <hip/hip_runtime.h>
#include <hip/hip_bf16.h>

// MI455X / gfx1250, wave32. All matmuls via v_wmma_f32_16x16x32_bf16 (f32 accum).
// D = N = C = 1024, B = 32 throughout. ~0.96 TFLOP of GEMM -> compute-bound,
// so: bf16 WMMA inner loops, async global->LDS staging, fp32 softmax/epilogues.

typedef __bf16 bf16_t;
typedef __attribute__((ext_vector_type(16))) __bf16       v16bf;
typedef __attribute__((ext_vector_type(8)))  float        v8f;
typedef __attribute__((ext_vector_type(4)))  unsigned int u32x4;
typedef __attribute__((ext_vector_type(2)))  unsigned int u32x2;
typedef __attribute__((ext_vector_type(4)))  float        f32x4;

#define DIMN 1024
#define BM   128
#define BN   128
#define BK   32

union Frag { v16bf v; u32x4 q[2]; bf16_t e[16]; };
union H8   { u32x4 q; bf16_t e[8]; };

// LDS byte-offset of a generic pointer into __shared__ (generic->AS3 cast).
__device__ __forceinline__ unsigned lds_addr_of(const void* p) {
  return (unsigned)(unsigned long long)(__attribute__((address_space(3))) const void*)p;
}
// CDNA5 async DMA: 16B per lane, global -> LDS, tracked by ASYNCcnt.
__device__ __forceinline__ void async_copy_b128(void* lds, const void* g) {
  asm volatile("global_load_async_to_lds_b128 %0, %1, off"
               :: "v"(lds_addr_of(lds)), "v"((unsigned long long)(size_t)g)
               : "memory");
}
__device__ __forceinline__ void wait_async0() {
  asm volatile("s_wait_asynccnt 0x0" ::: "memory");
}

// C[m,n] = alpha * sum_k opA(A)[m,k] * opB(B)[k,n]  (+bias, +resid)
// TA: A stored K x M (use A[k*1024+m]);  TB: B stored N x K (use B[n*1024+k])
// BIAS: 0 none, 1 per-row (bias[m]), 2 per-col (bias[n])
template<bool TA, bool TB, int BIAS, bool RESID, bool OF32, bool OBF16>
__global__ __launch_bounds__(256)
void gemm_bf16(const bf16_t* __restrict__ A,  long long sA,
               const bf16_t* __restrict__ Bm, long long sB,
               const float*  __restrict__ bias,
               const float*  __restrict__ resid, long long sR,
               float*        __restrict__ outf,  long long sOf,
               bf16_t*       __restrict__ outb,  long long sOb,
               float alpha)
{
  __shared__ __align__(16) bf16_t As[BM * BK];   // As[m][k]
  __shared__ __align__(16) bf16_t Bs[BN * BK];   // Bs[n][k]

  const int tid = threadIdx.x;
  const int bz  = blockIdx.z;
  const int m0  = blockIdx.y * BM;
  const int n0  = blockIdx.x * BN;
  const bf16_t* Ab = A  + (long long)bz * sA;
  const bf16_t* Bb = Bm + (long long)bz * sB;

  const int wid = tid >> 5, lane = tid & 31;
  const int wm  = wid & 3;      // 4 wave-rows  -> 32-row slab each
  const int wn  = wid >> 2;     // 2 wave-cols  -> 64-col slab each
  const int lh  = lane >> 4;    // half-wave (K split per ISA 7.12.2 layout)
  const int l15 = lane & 15;

  v8f acc[2][4] = {};

  for (int k0 = 0; k0 < DIMN; k0 += BK) {
    __syncthreads();
    // ---- stage A tile (BM x BK) into As[m][k] ----
    if (!TA) {
      int mi = tid >> 2;
      int kk = (tid & 3) * 8;
      async_copy_b128(&As[mi * BK + kk],
                      Ab + (long long)(m0 + mi) * DIMN + k0 + kk);
      async_copy_b128(&As[(mi + 64) * BK + kk],
                      Ab + (long long)(m0 + mi + 64) * DIMN + k0 + kk);
    } else {
      int kk = tid >> 3;
      int mi = (tid & 7) * 8;
      H8 d0, d1;
      d0.q = *(const u32x4*)(Ab + (long long)(k0 + kk) * DIMN + m0 + mi);
      d1.q = *(const u32x4*)(Ab + (long long)(k0 + kk) * DIMN + m0 + mi + 64);
      #pragma unroll
      for (int j = 0; j < 8; ++j) {
        As[(mi + j) * BK + kk]      = d0.e[j];
        As[(mi + 64 + j) * BK + kk] = d1.e[j];
      }
    }
    // ---- stage B tile (BK x BN) into Bs[n][k] ----
    if (!TB) {
      int kk = tid >> 3;
      int nb = (tid & 7) * 16;
      H8 d0, d1;
      d0.q = *(const u32x4*)(Bb + (long long)(k0 + kk) * DIMN + n0 + nb);
      d1.q = *(const u32x4*)(Bb + (long long)(k0 + kk) * DIMN + n0 + nb + 8);
      #pragma unroll
      for (int j = 0; j < 8; ++j) {
        Bs[(nb + j) * BK + kk]     = d0.e[j];
        Bs[(nb + 8 + j) * BK + kk] = d1.e[j];
      }
    } else {
      int ni = tid >> 1;
      int kk = (tid & 1) * 8;
      async_copy_b128(&Bs[ni * BK + kk],
                      Bb + (long long)(n0 + ni) * DIMN + k0 + kk);
      async_copy_b128(&Bs[ni * BK + kk + 16],
                      Bb + (long long)(n0 + ni) * DIMN + k0 + kk + 16);
    }
    if (k0 + BK < DIMN) {  // speculative prefetch of next tiles (global_prefetch_b8)
      __builtin_prefetch(Ab + (long long)(TA ? (k0 + BK) * DIMN + m0 : m0 * DIMN + k0 + BK), 0, 1);
      __builtin_prefetch(Bb + (long long)(TB ? n0 * DIMN + k0 + BK : (k0 + BK) * DIMN + n0), 0, 1);
    }
    if (!TA || TB) wait_async0();   // drain this wave's async LDS writes
    __syncthreads();

    // ---- fragments (ISA 7.12.2 wave32 layout) ----
    Frag a[2], b[4];
    #pragma unroll
    for (int i = 0; i < 2; ++i) {
      const int ar = (32 * wm + 16 * i + l15) * BK + 8 * lh;
      a[i].q[0] = *(const u32x4*)&As[ar];
      a[i].q[1] = *(const u32x4*)&As[ar + 16];
    }
    #pragma unroll
    for (int j = 0; j < 4; ++j) {
      const int br = (64 * wn + 16 * j + l15) * BK + 8 * lh;
      b[j].q[0] = *(const u32x4*)&Bs[br];
      b[j].q[1] = *(const u32x4*)&Bs[br + 16];
    }
    #pragma unroll
    for (int i = 0; i < 2; ++i)
      #pragma unroll
      for (int j = 0; j < 4; ++j)
        acc[i][j] = __builtin_amdgcn_wmma_f32_16x16x32_bf16(
            false, a[i].v, false, b[j].v, (short)0, acc[i][j], false, false);
  }

  // ---- epilogue: lane L, VGPR r holds M = r + 8*(L>=16), N = L&15 ----
  const int ma = m0 + 32 * wm + 8 * lh;
  const int nb = n0 + 64 * wn + l15;
  const float* Rb  = RESID ? (resid + (long long)bz * sR)  : nullptr;
  float*       Ofb = OF32  ? (outf  + (long long)bz * sOf) : nullptr;
  bf16_t*      Obb = OBF16 ? (outb  + (long long)bz * sOb) : nullptr;

  #pragma unroll
  for (int i = 0; i < 2; ++i) {
    #pragma unroll
    for (int r = 0; r < 8; ++r) {
      const int m = ma + 16 * i + r;
      const long long rowoff = (long long)m * DIMN;
      const float brow = (BIAS == 1) ? bias[m] : 0.0f;
      #pragma unroll
      for (int j = 0; j < 4; ++j) {
        const int n = nb + 16 * j;
        float v = acc[i][j][r] * alpha + brow;
        if (BIAS == 2) v += bias[n];
        const long long idx = rowoff + n;
        if (RESID) v += Rb[idx];
        if (OF32)  Ofb[idx] = v;
        if (OBF16) Obb[idx] = (bf16_t)v;
      }
    }
  }
}

// Row softmax over fp32 scores -> bf16 probabilities. One 256-thread block per row.
__global__ __launch_bounds__(256)
void softmax_rows(const float* __restrict__ S, bf16_t* __restrict__ P)
{
  __shared__ float red[256];
  const int t = threadIdx.x;
  const long long row = blockIdx.x;
  f32x4 v = ((const f32x4*)(S + row * DIMN))[t];
  float mx = fmaxf(fmaxf(v.x, v.y), fmaxf(v.z, v.w));
  red[t] = mx; __syncthreads();
  for (int o = 128; o > 0; o >>= 1) { if (t < o) red[t] = fmaxf(red[t], red[t + o]); __syncthreads(); }
  mx = red[0]; __syncthreads();
  float e0 = __expf(v.x - mx), e1 = __expf(v.y - mx);
  float e2 = __expf(v.z - mx), e3 = __expf(v.w - mx);
  red[t] = e0 + e1 + e2 + e3; __syncthreads();
  for (int o = 128; o > 0; o >>= 1) { if (t < o) red[t] += red[t + o]; __syncthreads(); }
  const float inv = 1.0f / red[0];
  union { bf16_t e[4]; u32x2 u; } pk;
  pk.e[0] = (bf16_t)(e0 * inv); pk.e[1] = (bf16_t)(e1 * inv);
  pk.e[2] = (bf16_t)(e2 * inv); pk.e[3] = (bf16_t)(e3 * inv);
  ((u32x2*)(P + row * DIMN))[t] = pk.u;
}

__global__ __launch_bounds__(256)
void cvt_f32_bf16(const float* __restrict__ in, bf16_t* __restrict__ out, int n4)
{
  int i = blockIdx.x * 256 + threadIdx.x;
  if (i >= n4) return;
  f32x4 v = ((const f32x4*)in)[i];
  union { bf16_t e[4]; u32x2 u; } pk;
  pk.e[0] = (bf16_t)v.x; pk.e[1] = (bf16_t)v.y;
  pk.e[2] = (bf16_t)v.z; pk.e[3] = (bf16_t)v.w;
  ((u32x2*)out)[i] = pk.u;
}

extern "C" void kernel_launch(void* const* d_in, const int* in_sizes, int n_in,
                              void* d_out, int out_size, void* d_ws, size_t ws_size,
                              hipStream_t stream)
{
  (void)in_sizes; (void)n_in; (void)out_size; (void)ws_size;
  const float* x = (const float*)d_in[0];
  const float* W[8]    = { (const float*)d_in[1],  (const float*)d_in[3],
                           (const float*)d_in[5],  (const float*)d_in[7],
                           (const float*)d_in[9],  (const float*)d_in[11],
                           (const float*)d_in[13], (const float*)d_in[15] }; // w1..w7, wl
  const float* bias[8] = { (const float*)d_in[2],  (const float*)d_in[4],
                           (const float*)d_in[6],  (const float*)d_in[8],
                           (const float*)d_in[10], (const float*)d_in[12],
                           (const float*)d_in[14], (const float*)d_in[16] }; // b1..b7, bl
  float* out = (float*)d_out;

  const long long SQ    = (long long)DIMN * DIMN;   // per-batch activation stride
  const size_t    ELEMS = (size_t)32 * (size_t)SQ;  // B*D*N

  char* ws = (char*)d_ws; size_t off = 0;
  auto take = [&](size_t bytes) -> void* {
    void* p = ws + off; off += (bytes + 255) & ~(size_t)255; return p;
  };
  bf16_t* Xb  = (bf16_t*)take(ELEMS * 2);  // bf16(x); reused as T/O2 scratch in block 2
  bf16_t* Qb  = (bf16_t*)take(ELEMS * 2);
  bf16_t* Kb  = (bf16_t*)take(ELEMS * 2);
  bf16_t* Vb  = (bf16_t*)take(ELEMS * 2);
  bf16_t* Pb  = (bf16_t*)take(ELEMS * 2);
  bf16_t* X1b = (bf16_t*)take(ELEMS * 2);
  float*  S   = (float*) take(ELEMS * 4);
  float*  X1f = (float*) take(ELEMS * 4);
  bf16_t* Wb[8];
  for (int i = 0; i < 8; ++i) Wb[i] = (bf16_t*)take((size_t)SQ * 2);

  const dim3 GB(256);
  const dim3 GG(DIMN / BN, DIMN / BM, 32);
  const float scale = 0.03125f;  // 1/sqrt(1024)

  // fp32 -> bf16 operand conversion
  cvt_f32_bf16<<<dim3((unsigned)(ELEMS / 4 / 256)), GB, 0, stream>>>(x, Xb, (int)(ELEMS / 4));
  for (int i = 0; i < 8; ++i)
    cvt_f32_bf16<<<dim3((unsigned)(SQ / 4 / 256)), GB, 0, stream>>>(W[i], Wb[i], (int)(SQ / 4));

  // ---------------- attention block 1 ----------------
  gemm_bf16<false,false,1,false,false,true ><<<GG,GB,0,stream>>>(Wb[0],0, Xb,SQ, bias[0], nullptr,0, nullptr,0, Qb,SQ, 1.0f);
  gemm_bf16<false,false,1,false,false,true ><<<GG,GB,0,stream>>>(Wb[1],0, Xb,SQ, bias[1], nullptr,0, nullptr,0, Kb,SQ, 1.0f);
  gemm_bf16<false,false,1,false,false,true ><<<GG,GB,0,stream>>>(Wb[2],0, Xb,SQ, bias[2], nullptr,0, nullptr,0, Vb,SQ, 1.0f);
  gemm_bf16<true ,false,0,false,true ,false><<<GG,GB,0,stream>>>(Qb,SQ, Kb,SQ, nullptr, nullptr,0, S,SQ, nullptr,0, scale);
  softmax_rows<<<dim3(32 * DIMN), GB, 0, stream>>>(S, Pb);
  gemm_bf16<false,true ,0,true ,true ,true ><<<GG,GB,0,stream>>>(Vb,SQ, Pb,SQ, nullptr, x,SQ, X1f,SQ, X1b,SQ, 1.0f);

  // ---------------- attention block 2 ----------------
  // Q2 = lin(conv(X1, w5, b5))
  gemm_bf16<false,false,1,false,false,true ><<<GG,GB,0,stream>>>(Wb[4],0, X1b,SQ, bias[4], nullptr,0, nullptr,0, Xb,SQ, 1.0f);
  gemm_bf16<false,true ,2,false,false,true ><<<GG,GB,0,stream>>>(Xb,SQ, Wb[7],0, bias[7], nullptr,0, nullptr,0, Qb,SQ, 1.0f);
  // K2 = lin(conv(X1, w6, b6))
  gemm_bf16<false,false,1,false,false,true ><<<GG,GB,0,stream>>>(Wb[5],0, X1b,SQ, bias[5], nullptr,0, nullptr,0, Xb,SQ, 1.0f);
  gemm_bf16<false,true ,2,false,false,true ><<<GG,GB,0,stream>>>(Xb,SQ, Wb[7],0, bias[7], nullptr,0, nullptr,0, Kb,SQ, 1.0f);
  // V2 = lin(conv(X1, w7, b7))
  gemm_bf16<false,false,1,false,false,true ><<<GG,GB,0,stream>>>(Wb[6],0, X1b,SQ, bias[6], nullptr,0, nullptr,0, Xb,SQ, 1.0f);
  gemm_bf16<false,true ,2,false,false,true ><<<GG,GB,0,stream>>>(Xb,SQ, Wb[7],0, bias[7], nullptr,0, nullptr,0, Vb,SQ, 1.0f);
  // scores, softmax, O2
  gemm_bf16<true ,false,0,false,true ,false><<<GG,GB,0,stream>>>(Qb,SQ, Kb,SQ, nullptr, nullptr,0, S,SQ, nullptr,0, scale);
  softmax_rows<<<dim3(32 * DIMN), GB, 0, stream>>>(S, Pb);
  gemm_bf16<false,true ,0,false,false,true ><<<GG,GB,0,stream>>>(Vb,SQ, Pb,SQ, nullptr, nullptr,0, nullptr,0, Xb,SQ, 1.0f);
  // X2 = conv(O2, w4, b4) + X1  -> fp32 output
  gemm_bf16<false,false,1,true ,true ,false><<<GG,GB,0,stream>>>(Wb[3],0, Xb,SQ, bias[3], X1f,SQ, out,SQ, nullptr,0, 1.0f);
}
